// Attention_1906965479775
// MI455X (gfx1250) — matrix operand order repaired and checked by reading, not compiled, not run on hardware
//
#include <hip/hip_runtime.h>
#include <math.h>

#define BB   64
#define SS   2048
#define ENCD 512
#define DECD 512

typedef __attribute__((ext_vector_type(16))) __bf16 v16bf;
typedef __attribute__((ext_vector_type(8)))  __bf16 v8bf;
typedef __attribute__((ext_vector_type(8)))  float  v8f;

__device__ __forceinline__ unsigned int f2bf(float f) {
    __bf16 h = (__bf16)f;
    return (unsigned int)__builtin_bit_cast(unsigned short, h);
}

// Branchless tanh: t = exp(2x); tanh = 1 - 2/(t+1).  ~5 VALU ops, saturates
// correctly (t->inf => 1, t->0 => -1).  No divergent slow path.
__device__ __forceinline__ float fast_tanh(float x) {
    float t = __expf(2.0f * x);
    return 1.0f - 2.0f * __builtin_amdgcn_rcpf(t + 1.0f);
}

// ---------------------------------------------------------------------------
// Kernel 1: repack W_e (rows DECD.. of W_attn) into bf16 WMMA B-fragment layout
// frag(kt,nt) -> 32 lanes x 32 B.  (ISA 7.12.2, B 32x16 bf16):
//   VGPR v, lanes 0-15 : {B[k,n], B[k+1,n]}, k = 2(v&3)+16(v>>2), n = lane
//   VGPR v, lanes 16-31: {B[k+8,n], B[k+9,n]}, same k, n = lane-16
// ---------------------------------------------------------------------------
__global__ __launch_bounds__(256) void pack_we(const float* __restrict__ Wattn,
                                               unsigned int* __restrict__ wep) {
    int t    = blockIdx.x * 256 + threadIdx.x;   // 0 .. 131071
    int v    = t & 7;
    int lane = (t >> 3) & 31;
    int nt   = (t >> 8) & 31;
    int kt   = t >> 13;
    int n = nt * 16 + (lane & 15);
    int k = kt * 32 + ((lane >> 4) ? 8 : 0) + 2 * (v & 3) + ((v >> 2) ? 16 : 0);
    float w0 = Wattn[(size_t)(DECD + k) * DECD + n];
    float w1 = Wattn[(size_t)(DECD + k + 1) * DECD + n];
    wep[(((kt * 32 + nt) * 32) + lane) * 8 + v] = f2bf(w0) | (f2bf(w1) << 16);
}

// ---------------------------------------------------------------------------
// Kernel 2: hp[b][d] = hidden[b,:] @ W_h + b_attn[d]   (tiny)
// ---------------------------------------------------------------------------
__global__ __launch_bounds__(256) void hp_prep(const float* __restrict__ hidden,
                                               const float* __restrict__ Wattn,
                                               const float* __restrict__ bias,
                                               float* __restrict__ hp) {
    __shared__ float h[DECD];
    int b = blockIdx.x;
    int d = blockIdx.y * 256 + threadIdx.x;
    h[threadIdx.x]       = hidden[b * DECD + threadIdx.x];
    h[threadIdx.x + 256] = hidden[b * DECD + threadIdx.x + 256];
    __syncthreads();
    float acc = bias[d];
#pragma unroll 8
    for (int k = 0; k < DECD; k++)
        acc += h[k] * Wattn[(size_t)k * DECD + d];
    hp[b * DECD + d] = acc;
}

// ---------------------------------------------------------------------------
// Kernel 3: main WMMA kernel.  Block = (b, 64-row s-tile); grid = 2048 x 256thr.
// A staged in LDS as bf16, stride 1024 B, XOR-swizzled 16B chunks
// (chunk ^ (row&15)) -> conflict-free ds_load_b128 A-fragments; exactly 64 KB.
// Per wave: 4 m-tiles x 4 n-tiles, K-loop 16 -> 256 v_wmma per wave.
// ---------------------------------------------------------------------------
__global__ __launch_bounds__(256) void energy_scores(
        const float* __restrict__ enc, const unsigned int* __restrict__ wep,
        const float* __restrict__ hp,  const float* __restrict__ wv,
        float* __restrict__ scores) {
    __shared__ unsigned short Asm[64 * 512];    // 65536 B exactly

    const int tid = threadIdx.x;
    const int b   = blockIdx.x >> 5;
    const int st  = blockIdx.x & 31;

    // ---- Stage enc[b, st*64 .. +64, :] as bf16 into swizzled LDS ----
    const float4* e4 = (const float4*)(enc + ((size_t)b * SS + st * 64) * ENCD);
#pragma unroll
    for (int i = 0; i < 32; i++) {
        int f   = tid + i * 256;        // float4 index, 0..8191
        int row = f >> 7;               // 128 float4 per row
        int c4  = f & 127;              // 8-byte piece within row
        float4 vv = e4[(size_t)row * 128 + c4];
        uint2 p;
        p.x = f2bf(vv.x) | (f2bf(vv.y) << 16);
        p.y = f2bf(vv.z) | (f2bf(vv.w) << 16);
        int ch = (c4 >> 1) ^ (row & 15);                    // swizzled 16B chunk
        *(uint2*)&Asm[row * 512 + ch * 8 + (c4 & 1) * 4] = p;
    }
    __syncthreads();

    const int w    = tid >> 5;
    const int lane = tid & 31;
    const int half = lane >> 4;
    const int mrow = lane & 15;

    v8f acc[4][4];
    v8f zero = {};
#pragma unroll
    for (int mi = 0; mi < 4; mi++)
#pragma unroll
        for (int ni = 0; ni < 4; ni++) acc[mi][ni] = zero;

    const v8bf* wp = (const v8bf*)wep;

    for (int kt = 0; kt < 16; kt++) {
        v16bf aF[4], bF[4];
        // A fragments (16x32 bf16): lanes0-15 rows, K 0-7 & 16-23; lanes16-31 +8
#pragma unroll
        for (int mi = 0; mi < 4; mi++) {
            int row = mi * 16 + mrow;            // row&15 == mrow
            int c0  = kt * 4 + half;             // 16B chunk of first load
            v8bf lo = *(const v8bf*)&Asm[row * 512 + ((c0)     ^ mrow) * 8];
            v8bf hi = *(const v8bf*)&Asm[row * 512 + ((c0 + 2) ^ mrow) * 8];
            aF[mi] = __builtin_shufflevector(lo, hi,
                       0,1,2,3,4,5,6,7,8,9,10,11,12,13,14,15);
        }
        // B fragments from pre-packed buffer: 2 x global_load_b128 per lane
#pragma unroll
        for (int ni = 0; ni < 4; ni++) {
            int nt   = w * 4 + ni;
            int base = ((kt * 32 + nt) * 32 + lane) * 2;
            v8bf lo = wp[base];
            v8bf hi = wp[base + 1];
            bF[ni] = __builtin_shufflevector(lo, hi,
                       0,1,2,3,4,5,6,7,8,9,10,11,12,13,14,15);
        }
#pragma unroll
        for (int mi = 0; mi < 4; mi++)
#pragma unroll
            for (int ni = 0; ni < 4; ni++)
                acc[mi][ni] = __builtin_amdgcn_wmma_f32_16x16x32_bf16(
                    false, aF[mi], false, bF[ni], (short)0, acc[mi][ni],
                    false, false);
    }

    // ---- Epilogue: +hp, tanh, .w_v ; per-lane partial over its 4 n values ----
    float hpv[4], wvv[4];
#pragma unroll
    for (int ni = 0; ni < 4; ni++) {
        int n = (w * 4 + ni) * 16 + mrow;
        hpv[ni] = hp[b * DECD + n];
        wvv[ni] = wv[n];
    }
    // C/D layout: element r of v8f -> M = r + 8*half, N = mrow
    float sBuf[4][8];
#pragma unroll
    for (int mi = 0; mi < 4; mi++) {
#pragma unroll
        for (int r = 0; r < 8; r++) {
            float s = 0.f;
#pragma unroll
            for (int ni = 0; ni < 4; ni++)
                s += wvv[ni] * fast_tanh(acc[mi][ni][r] + hpv[ni]);
            sBuf[mi][r] = s;
        }
    }
    // ---- Cross-lane + cross-wave reduction through (reused) LDS ----
    // Layout: row = (w*2 + half)*16 + mrow, stride 34 dwords (pad -> distinct
    // even bank starts for ds_store_b64 across all 32 lanes; conflict-free).
    __syncthreads();                 // Asm dead everywhere; reuse it
    float* red2 = (float*)Asm;       // 256 rows x 34 floats = 34 KB
    {
        int rowb = ((w * 2 + half) * 16 + mrow) * 34;
#pragma unroll
        for (int mi = 0; mi < 4; mi++)
#pragma unroll
            for (int j = 0; j < 4; j++) {
                float2 p = make_float2(sBuf[mi][2 * j], sBuf[mi][2 * j + 1]);
                *(float2*)&red2[rowb + mi * 8 + 2 * j] = p;
            }
    }
    __syncthreads();
    if (tid < 64) {
        // tid == m = mi*16 + half*8 + r
        int mi = tid >> 4, hf = (tid >> 3) & 1, r = tid & 7;
        int slot = mi * 8 + r;
        float sc = 0.f;
#pragma unroll
        for (int q = 0; q < 8; q++)          // waves
#pragma unroll
            for (int mr = 0; mr < 16; mr++)  // lanes of the matching half
                sc += red2[((q * 2 + hf) * 16 + mr) * 34 + slot];
        scores[(size_t)b * SS + st * 64 + tid] = sc;
    }
}

// ---------------------------------------------------------------------------
// Kernel 4: softmax over S per batch row (in place)
// ---------------------------------------------------------------------------
__global__ __launch_bounds__(256) void softmax_k(float* __restrict__ sc) {
    __shared__ float red[256];
    int b = blockIdx.x, tid = threadIdx.x;
    float v[8];
    float mx = -1e30f;
#pragma unroll
    for (int i = 0; i < 8; i++) {
        v[i] = sc[(size_t)b * SS + tid + i * 256];
        mx = fmaxf(mx, v[i]);
    }
    red[tid] = mx; __syncthreads();
    for (int off = 128; off > 0; off >>= 1) {
        if (tid < off) red[tid] = fmaxf(red[tid], red[tid + off]);
        __syncthreads();
    }
    mx = red[0]; __syncthreads();
    float sum = 0.f;
#pragma unroll
    for (int i = 0; i < 8; i++) { v[i] = __expf(v[i] - mx); sum += v[i]; }
    red[tid] = sum; __syncthreads();
    for (int off = 128; off > 0; off >>= 1) {
        if (tid < off) red[tid] += red[tid + off];
        __syncthreads();
    }
    float inv = 1.0f / red[0];
#pragma unroll
    for (int i = 0; i < 8; i++)
        sc[(size_t)b * SS + tid + i * 256] = v[i] * inv;
}

// ---------------------------------------------------------------------------
// Kernel 5: partial weighted sums, grid (64,16) x 128 thr, 128 s per chunk
// ---------------------------------------------------------------------------
__global__ __launch_bounds__(128) void wsum_k(const float* __restrict__ enc,
                                              const float* __restrict__ attn,
                                              float* __restrict__ partial) {
    int b = blockIdx.x, sc = blockIdx.y, t = threadIdx.x;
    const float4* e4 = (const float4*)enc;
    float4 acc = make_float4(0.f, 0.f, 0.f, 0.f);
    int s0 = sc * 128;
#pragma unroll 4
    for (int s = 0; s < 128; s++) {
        int ss = s0 + s;
        float a = attn[(size_t)b * SS + ss];
        float4 vv = e4[(((size_t)b * SS + ss) * ENCD >> 2) + t];
        acc.x += a * vv.x; acc.y += a * vv.y;
        acc.z += a * vv.z; acc.w += a * vv.w;
    }
    ((float4*)partial)[(size_t)(b * 16 + sc) * (ENCD >> 2) + t] = acc;
}

// Kernel 6: out[b][e] = sum over 16 s-chunk partials (fixed order)
__global__ __launch_bounds__(256) void reduce_out(const float* __restrict__ partial,
                                                  float* __restrict__ out) {
    int idx = blockIdx.x * 256 + threadIdx.x;   // 0..32767
    int b = idx >> 9, e = idx & 511;
    float s = 0.f;
#pragma unroll
    for (int q = 0; q < 16; q++) s += partial[(size_t)((b * 16 + q) << 9) + e];
    out[idx] = s;
}

extern "C" void kernel_launch(void* const* d_in, const int* in_sizes, int n_in,
                              void* d_out, int out_size, void* d_ws, size_t ws_size,
                              hipStream_t stream) {
    const float* hidden = (const float*)d_in[0];
    const float* enc    = (const float*)d_in[1];
    const float* Wattn  = (const float*)d_in[2];
    const float* bias   = (const float*)d_in[3];
    const float* wv     = (const float*)d_in[4];
    float* out = (float*)d_out;

    char* ws = (char*)d_ws;
    float*        hp      = (float*)ws;                               // 128 KB
    unsigned int* wep     = (unsigned int*)(ws + 131072);             // 512 KB
    float*        scores  = (float*)(ws + 131072 + 524288);           // 512 KB
    float*        partial = (float*)(ws + 131072 + 524288 + 524288);  // 2 MB

    pack_we      <<<512,          256, 0, stream>>>(Wattn, wep);
    hp_prep      <<<dim3(64, 2),  256, 0, stream>>>(hidden, Wattn, bias, hp);
    energy_scores<<<2048,         256, 0, stream>>>(enc, wep, hp, wv, scores);
    softmax_k    <<<64,           256, 0, stream>>>(scores);
    wsum_k       <<<dim3(64, 16), 128, 0, stream>>>(enc, scores, partial);
    reduce_out   <<<128,          256, 0, stream>>>(partial, out);
}
